// Attention_7009386627377
// MI455X (gfx1250) — matrix operand order repaired and checked by reading, not compiled, not run on hardware
//
#include <hip/hip_runtime.h>

// ---------------------------------------------------------------------------
// CDNA5 (gfx1250, wave32) fused attention:
//   QKV projection GEMMs + flash-attention, matmuls via
//   v_wmma_f32_16x16x32_bf16 (fp32 accumulate), softmax in fp32.
//   Shared tiles staged through LDS with global_load_async_to_lds_b128
//   (double buffered, ASYNCcnt-synchronized). Fragments are prefetched in
//   batches so WMMAs issue back-to-back behind a single dscnt wait.
// ---------------------------------------------------------------------------

typedef __attribute__((ext_vector_type(16))) __bf16 v16bf;
typedef __attribute__((ext_vector_type(8)))  float  v8f;
typedef __attribute__((ext_vector_type(4)))  unsigned int u32x4;
typedef __attribute__((ext_vector_type(4)))  int i32x4;

#define HID   1024
#define NH    16
#define DH    64
#define BBATCH 4
#define SSEQ  2048
#define ROWS  (BBATCH * SSEQ)   // 8192
#define MASK_MIN (-3.4028234663852886e38f)

#if defined(__has_builtin)
#  if __has_builtin(__builtin_amdgcn_global_load_async_to_lds_b128)
#    define HAVE_ASYNC_LDS 1
#  endif
#endif
#ifndef HAVE_ASYNC_LDS
#  define HAVE_ASYNC_LDS 0
#endif

// Wait on this wave's outstanding async LDS copies (ASYNCcnt <= N).
#define WAIT_ASYNC(N) asm volatile("s_wait_asynccnt %0" ::"i"(N) : "memory")

// One 16-byte global -> LDS copy per lane.
__device__ __forceinline__ void async_cp16(const __bf16* g, __bf16* l)
{
#if HAVE_ASYNC_LDS
    // Builtin signature (probe-confirmed): (v4i AS1*, v4i AS3*, Ii, Ii)
    typedef __attribute__((address_space(1))) i32x4 gvec_t;
    typedef __attribute__((address_space(3))) i32x4 lvec_t;
    __builtin_amdgcn_global_load_async_to_lds_b128(
        (gvec_t*)g, (lvec_t*)l, 0, 0);
#else
    *(u32x4*)l = *(const u32x4*)g;   // fallback: load + ds_store_b128
#endif
}

// 32-byte fragment: 16 bf16 elements = one WMMA A or B operand per lane.
union Frag {
    u32x4 u[2];
    v16bf v;
};

__device__ __forceinline__ void load_frag(Frag& f, const __bf16* p0,
                                          const __bf16* p1)
{
    f.u[0] = *(const u32x4*)p0;
    f.u[1] = *(const u32x4*)p1;
}

// ---------------------------------------------------------------------------
// fp32 -> bf16 elementwise
// ---------------------------------------------------------------------------
__global__ __launch_bounds__(256) void cvt_f32_bf16(
    const float* __restrict__ in, __bf16* __restrict__ out, int n)
{
    int i = blockIdx.x * blockDim.x + threadIdx.x;
    if (i < n) out[i] = (__bf16)in[i];
}

// ---------------------------------------------------------------------------
// W [K=1024, N=1024] fp32  ->  Wt [N, K] bf16  (transpose so B-fragments are
// contiguous along K per the ISA 16-bit B-matrix layout)
// ---------------------------------------------------------------------------
__global__ __launch_bounds__(256) void cvt_transpose_w(
    const float* __restrict__ in, __bf16* __restrict__ out)
{
    int i = blockIdx.x * blockDim.x + threadIdx.x;   // over 1024*1024
    int n = i >> 10;
    int k = i & 1023;
    out[i] = (__bf16)in[k * HID + n];                // out[n][k] = in[k][n]
}

// ---------------------------------------------------------------------------
// GEMM: [8192 x 1024](bf16) x Wt[1024 x 1024](bf16, pre-transposed) + bias
// Each wave: 16(M) x 64(N) tile, 4 v8f accumulators. 8 waves/WG -> 128x64.
// The shared 64x32 W tile is staged through double-buffered LDS with
// async-to-LDS copies (one b128 per thread per k-step).
// vmode==0 : dst[((b*NH+h)*SSEQ + s)*DH + d]     (Q, K layout)
// vmode==1 : dst[((b*NH+h)*DH + d)*SSEQ + s]     (V transposed)
// ---------------------------------------------------------------------------
__global__ __launch_bounds__(256) void qkv_gemm(
    const __bf16* __restrict__ X, const __bf16* __restrict__ Wt,
    const float* __restrict__ bias, __bf16* __restrict__ dst, int vmode)
{
    __shared__ __bf16 wtile[2][64 * 32];   // [buf][n*32 + k], 4 KB each

    const int lane = threadIdx.x & 31;
    const int wave = threadIdx.x >> 5;
    const int half = lane >> 4;
    const int l15  = lane & 15;
    const int m0   = blockIdx.x * 128 + wave * 16;
    const int n0   = blockIdx.y * 64;

    // staging indices: thread t copies row n = t/4, cols kc..kc+7 (16 B)
    const int sn = threadIdx.x >> 2;
    const int sk = (threadIdx.x & 3) << 3;

    auto stage_w = [&](int buf, int k0) {
        async_cp16(Wt + (size_t)(n0 + sn) * HID + k0 + sk,
                   &wtile[buf][sn * 32 + sk]);
    };

    v8f acc[4] = {};

    stage_w(0, 0);
    for (int k0 = 0; k0 < HID; k0 += 32) {
        const int  buf     = (k0 >> 5) & 1;
        const bool hasnext = (k0 + 32) < HID;
        if (hasnext) { stage_w(buf ^ 1, k0 + 32); WAIT_ASYNC(1); }
        else         { WAIT_ASYNC(0); }
        __syncthreads();   // tile[buf] visible to all waves

        // A fragment (global): row M = m0 + (lane&15); K split per lane half
        Frag a;
        const __bf16* apx = X + (size_t)(m0 + l15) * HID + k0 + half * 8;
        load_frag(a, apx, apx + 16);

        // Prefetch all 4 B fragments, then issue the 4 WMMAs back-to-back.
        Frag bfr[4];
#pragma unroll
        for (int nt = 0; nt < 4; ++nt) {
            const __bf16* bp = &wtile[buf][(nt * 16 + l15) * 32 + half * 8];
            load_frag(bfr[nt], bp, bp + 16);
        }
#pragma unroll
        for (int nt = 0; nt < 4; ++nt)
            acc[nt] = __builtin_amdgcn_wmma_f32_16x16x32_bf16(
                false, a.v, false, bfr[nt].v, (short)0, acc[nt], false, false);

        __syncthreads();   // safe to overwrite tile[buf] next-next step
    }

    // Epilogue: C layout is M = r + 8*half, N = lane&15 (per VGPR r)
#pragma unroll
    for (int nt = 0; nt < 4; ++nt) {
#pragma unroll
        for (int r = 0; r < 8; ++r) {
            int gm = m0 + r + 8 * half;           // global row in [0,8192)
            int gc = n0 + nt * 16 + l15;          // global col in [0,1024)
            float val = acc[nt][r] + bias[gc];
            int b = gm >> 11, s = gm & 2047;
            int h = gc >> 6,  d = gc & 63;
            size_t idx;
            if (vmode) idx = (((size_t)(b * NH + h)) * DH + d) * SSEQ + s;
            else       idx = (((size_t)(b * NH + h)) * SSEQ + s) * DH + d;
            dst[idx] = (__bf16)val;
        }
    }
}

// ---------------------------------------------------------------------------
// Flash attention. One wave handles 16 query rows of one (b,h);
// 8 waves/WG -> 128 query rows per block. Key loop in chunks of 64.
// K and V chunks (8 KB each) are shared by all 8 waves and staged through
// double-buffered LDS with async-to-LDS copies (4 b128 per thread per chunk).
// ---------------------------------------------------------------------------
__global__ __launch_bounds__(256) void flash_attn(
    const __bf16* __restrict__ Q, const __bf16* __restrict__ K,
    const __bf16* __restrict__ Vt, const float* __restrict__ amask,
    float* __restrict__ out)
{
    __shared__ __bf16 ktile[2][64 * 64];   // [buf][key*64 + d], 8 KB each
    __shared__ __bf16 vtile[2][64 * 64];   // [buf][d*64 + key], 8 KB each
    __shared__ __bf16 plds[8][16 * 64];    // per-wave P staging, 2 KB each

    const int lane = threadIdx.x & 31;
    const int wave = threadIdx.x >> 5;
    const int half = lane >> 4;
    const int l15  = lane & 15;

    const int bh = blockIdx.x >> 4;       // 64 (b,h) pairs
    const int qt = blockIdx.x & 15;       // 16 q-tiles of 128
    const int b  = bh >> 4;
    const int h  = bh & 15;
    const int q0 = qt * 128 + wave * 16;

    const __bf16* Qh = Q  + (size_t)bh * SSEQ * DH;
    const __bf16* Kh = K  + (size_t)bh * SSEQ * DH;
    const __bf16* Vh = Vt + (size_t)bh * DH * SSEQ;
    const float*  mk = amask + b * SSEQ;

    auto stage_kv = [&](int buf, int kb) {
#pragma unroll
        for (int j = 0; j < 2; ++j) {
            int flat = threadIdx.x + j * 256;        // 0..511
            int row  = flat >> 3;                    // 0..63
            int col  = (flat & 7) << 3;              // 0,8,..,56
            // K tile: [key][d], same layout as global
            async_cp16(Kh + (size_t)(kb + row) * DH + col,
                       &ktile[buf][row * 64 + col]);
            // V tile: [d][key], from transposed global Vt
            async_cp16(Vh + (size_t)row * SSEQ + kb + col,
                       &vtile[buf][row * 64 + col]);
        }
    };

    // Q A-fragments for full K-dim (d = 64 -> two 16x16x32 k-steps)
    Frag aq[2];
#pragma unroll
    for (int kk = 0; kk < 2; ++kk) {
        const __bf16* qp = Qh + (size_t)(q0 + l15) * DH + kk * 32 + half * 8;
        load_frag(aq[kk], qp, qp + 16);
    }

    v8f ctx[4] = {};
    float mrow[8], lrow[8];
#pragma unroll
    for (int r = 0; r < 8; ++r) { mrow[r] = -1e30f; lrow[r] = 0.0f; }

    const float scale = 0.125f;   // 1/sqrt(64)

    stage_kv(0, 0);
    for (int c = 0; c < SSEQ / 64; ++c) {
        const int  kb      = c * 64;
        const int  buf     = c & 1;
        const bool hasnext = (c + 1) < (SSEQ / 64);
        if (hasnext) { stage_kv(buf ^ 1, kb + 64); WAIT_ASYNC(4); }
        else         { WAIT_ASYNC(0); }
        __syncthreads();   // K/V tile[buf] visible to all waves

        // ---- scores S = Q K^T : 16 x 64 (4 accumulators) ----
        v8f sc[4] = {};
#pragma unroll
        for (int kk = 0; kk < 2; ++kk) {
            // Prefetch 4 K^T B-fragments, then 4 WMMAs back-to-back.
            Frag bk[4];
#pragma unroll
            for (int nt = 0; nt < 4; ++nt) {
                const __bf16* kp = &ktile[buf][(nt * 16 + l15) * 64
                                               + kk * 32 + half * 8];
                load_frag(bk[nt], kp, kp + 16);
            }
#pragma unroll
            for (int nt = 0; nt < 4; ++nt)
                sc[nt] = __builtin_amdgcn_wmma_f32_16x16x32_bf16(
                    false, aq[kk].v, false, bk[nt].v, (short)0, sc[nt],
                    false, false);
        }

        // ---- scale + additive mask (column = key position) ----
        float madd[4];
#pragma unroll
        for (int nt = 0; nt < 4; ++nt)
            madd[nt] = (1.0f - mk[kb + nt * 16 + l15]) * MASK_MIN;
#pragma unroll
        for (int nt = 0; nt < 4; ++nt)
#pragma unroll
            for (int r = 0; r < 8; ++r)
                sc[nt][r] = sc[nt][r] * scale + madd[nt];

        // ---- online softmax per row (row = r + 8*half) ----
#pragma unroll
        for (int r = 0; r < 8; ++r) {
            float t = fmaxf(fmaxf(sc[0][r], sc[1][r]),
                            fmaxf(sc[2][r], sc[3][r]));
#pragma unroll
            for (int xm = 8; xm >= 1; xm >>= 1)
                t = fmaxf(t, __shfl_xor(t, xm, 32));  // stays inside 16-lane half
            float mnew  = fmaxf(mrow[r], t);
            float alpha = __expf(mrow[r] - mnew);
            float rs = 0.0f;
#pragma unroll
            for (int nt = 0; nt < 4; ++nt) {
                float p = __expf(sc[nt][r] - mnew);
                sc[nt][r] = p;
                rs += p;
            }
#pragma unroll
            for (int xm = 8; xm >= 1; xm >>= 1)
                rs += __shfl_xor(rs, xm, 32);
            lrow[r] = lrow[r] * alpha + rs;
            mrow[r] = mnew;
#pragma unroll
            for (int nt = 0; nt < 4; ++nt)
                ctx[nt][r] *= alpha;
        }

        // ---- P: accumulator layout -> A layout via per-wave LDS ----
        __bf16* pw = &plds[wave][0];
#pragma unroll
        for (int nt = 0; nt < 4; ++nt)
#pragma unroll
            for (int r = 0; r < 8; ++r)
                pw[(r + 8 * half) * 64 + nt * 16 + l15] = (__bf16)sc[nt][r];
        // same-wave LDS RAW: compiler inserts s_wait_dscnt

        // ---- ctx += P V : K = 64 keys (2 k-steps), N = dv (4 tiles) ----
#pragma unroll
        for (int kk = 0; kk < 2; ++kk) {
            Frag ap;
            const __bf16* pp = pw + (size_t)l15 * 64 + kk * 32 + half * 8;
            load_frag(ap, pp, pp + 16);

            // Prefetch 4 V B-fragments, then 4 WMMAs back-to-back.
            Frag bv[4];
#pragma unroll
            for (int nt = 0; nt < 4; ++nt) {
                const __bf16* vp = &vtile[buf][(nt * 16 + l15) * 64
                                               + kk * 32 + half * 8];
                load_frag(bv[nt], vp, vp + 16);
            }
#pragma unroll
            for (int nt = 0; nt < 4; ++nt)
                ctx[nt] = __builtin_amdgcn_wmma_f32_16x16x32_bf16(
                    false, ap.v, false, bv[nt].v, (short)0, ctx[nt],
                    false, false);
        }
        __syncthreads();   // safe to overwrite K/V tile[buf]
    }

    // ---- output [B, S, H, D] fp32, normalized by row sum ----
#pragma unroll
    for (int nt = 0; nt < 4; ++nt)
#pragma unroll
        for (int r = 0; r < 8; ++r) {
            int s = q0 + r + 8 * half;
            int d = nt * 16 + l15;
            out[(((size_t)b * SSEQ + s) * NH + h) * DH + d] =
                ctx[nt][r] / lrow[r];
        }
}

// ---------------------------------------------------------------------------
// Host-side launch
// ---------------------------------------------------------------------------
extern "C" void kernel_launch(void* const* d_in, const int* in_sizes, int n_in,
                              void* d_out, int out_size, void* d_ws, size_t ws_size,
                              hipStream_t stream)
{
    const float* X   = (const float*)d_in[0];
    const float* msk = (const float*)d_in[1];
    const float* Wq  = (const float*)d_in[2];
    const float* bq  = (const float*)d_in[3];
    const float* Wk  = (const float*)d_in[4];
    const float* bk  = (const float*)d_in[5];
    const float* Wv  = (const float*)d_in[6];
    const float* bv  = (const float*)d_in[7];
    float* out = (float*)d_out;

    // Workspace layout (~70 MB total, all bf16 intermediates)
    char* ws = (char*)d_ws;
    size_t off = 0;
    __bf16* Xbf = (__bf16*)(ws + off); off += (size_t)ROWS * HID * 2;
    __bf16* WtQ = (__bf16*)(ws + off); off += (size_t)HID * HID * 2;
    __bf16* WtK = (__bf16*)(ws + off); off += (size_t)HID * HID * 2;
    __bf16* WtV = (__bf16*)(ws + off); off += (size_t)HID * HID * 2;
    __bf16* Qb  = (__bf16*)(ws + off); off += (size_t)ROWS * HID * 2;
    __bf16* Kb  = (__bf16*)(ws + off); off += (size_t)ROWS * HID * 2;
    __bf16* Vtb = (__bf16*)(ws + off); off += (size_t)ROWS * HID * 2;
    (void)ws_size; (void)in_sizes; (void)n_in; (void)out_size;

    // 1) downconvert activations + transpose/downconvert weights
    {
        int n = ROWS * HID;
        cvt_f32_bf16<<<(n + 255) / 256, 256, 0, stream>>>(X, Xbf, n);
        int nw = HID * HID;
        cvt_transpose_w<<<(nw + 255) / 256, 256, 0, stream>>>(Wq, WtQ);
        cvt_transpose_w<<<(nw + 255) / 256, 256, 0, stream>>>(Wk, WtK);
        cvt_transpose_w<<<(nw + 255) / 256, 256, 0, stream>>>(Wv, WtV);
    }

    // 2) QKV projections (WMMA bf16, fp32 accumulate)
    {
        dim3 grid(ROWS / 128, HID / 64);   // 64 x 16
        qkv_gemm<<<grid, 256, 0, stream>>>(Xbf, WtQ, bq, Qb, 0);
        qkv_gemm<<<grid, 256, 0, stream>>>(Xbf, WtK, bk, Kb, 0);
        qkv_gemm<<<grid, 256, 0, stream>>>(Xbf, WtV, bv, Vtb, 1);
    }

    // 3) flash attention
    flash_attn<<<BBATCH * NH * (SSEQ / 128), 256, 0, stream>>>(
        Qb, Kb, Vtb, msk, out);
}
